// FSUMGUCell_29540785062379
// MI455X (gfx1250) — matrix operand order repaired and checked by reading, not compiled, not run on hardware
//
#include <hip/hip_runtime.h>
#include <stdint.h>

#define BATCH 2048
#define HID   2048
#define INPW  2048
#define KTOT  (HID + INPW)   // 4096

#define BM 256               // M rows per block
#define BN 128               // N cols per block
#define BK 32                // K per step
#define NTHREADS 256         // 8 waves
#define LDS_PITCH 20         // uints per 32-bf16 row (64B data + 16B pad): conflict-free

typedef __bf16 bf16_t;
typedef bf16_t v16bf __attribute__((ext_vector_type(16)));
typedef float  v8f   __attribute__((ext_vector_type(8)));

union FragU { uint4 u[2]; v16bf v; };

__device__ __forceinline__ uint32_t pack2_bf16(float lo, float hi) {
  uint32_t a = __builtin_bit_cast(uint32_t, lo);
  uint32_t b = __builtin_bit_cast(uint32_t, hi);
  a += 0x7FFFu + ((a >> 16) & 1u);          // round-to-nearest-even
  b += 0x7FFFu + ((b >> 16) & 1u);
  return (a >> 16) | (b & 0xFFFF0000u);
}

__device__ __forceinline__ uint16_t f32_to_bf16(float x) {
  uint32_t a = __builtin_bit_cast(uint32_t, x);
  a += 0x7FFFu + ((a >> 16) & 1u);
  return (uint16_t)(a >> 16);
}

// Async DMA: global (64-bit vaddr) -> LDS (32-bit lds addr); IOFFSET applies
// to BOTH addresses (ISA async-load pseudocode). Tracked by ASYNCcnt.
#define ASYNC_B128(lds, gptr, imm)                                          \
  asm volatile("global_load_async_to_lds_b128 %0, %1, off offset:" #imm    \
               :: "v"(lds), "v"(gptr) : "memory")

// ---------------------------------------------------------------------------
// Prep: fp32 -> bf16 bulk conversion (memory-bound), 8 elems/thread
// ---------------------------------------------------------------------------
__global__ __launch_bounds__(256)
void mgu_cvt_bf16(const float* __restrict__ src, uint16_t* __restrict__ dst) {
  size_t base = ((size_t)blockIdx.x * blockDim.x + threadIdx.x) * 8;
  const float4* s4 = reinterpret_cast<const float4*>(src + base);
  float4 f0 = s4[0], f1 = s4[1];
  uint4 o = make_uint4(pack2_bf16(f0.x, f0.y), pack2_bf16(f0.z, f0.w),
                       pack2_bf16(f1.x, f1.y), pack2_bf16(f1.z, f1.w));
  *reinterpret_cast<uint4*>(dst + base) = o;
}

// ---------------------------------------------------------------------------
// bf16 WMMA GEMM: C[BATCH x HID] = [aL | aR] @ w16^T, double-buffered LDS,
// async global->LDS DMA. Epilogue differs per phase.
// ---------------------------------------------------------------------------
template <int PHASE>
__global__ __launch_bounds__(NTHREADS, 1)
void mgu_gemm(const uint16_t* __restrict__ aL,   // k <  HID  (hx16 or fgh16)
              const uint16_t* __restrict__ aR,   // k >= HID  (in16)
              const uint16_t* __restrict__ w16,  // weights bf16 [HID x KTOT]
              const float* __restrict__ bias_v,
              const float* __restrict__ hx,
              float* __restrict__ fg_ws,
              uint16_t* __restrict__ fgh16,
              float* __restrict__ out)
{
  __shared__ __align__(16) uint32_t As[2][BM * LDS_PITCH];  // 2 x 20 KB
  __shared__ __align__(16) uint32_t Bs[2][BN * LDS_PITCH];  // 2 x 10 KB

  const int tid   = threadIdx.x;
  const int lane  = tid & 31;
  const int wave  = tid >> 5;
  const int waveM = (wave >> 1) * 64;     // 4 wave-rows x 64
  const int waveN = (wave & 1) * 64;      // 2 wave-cols x 64
  const int blockN = blockIdx.x * BN;
  const int blockM = blockIdx.y * BM;
  const int fr = lane & 15;               // fragment row/col
  const int fh = lane >> 4;               // fragment K half
  const int brow = tid >> 1;              // B loader row 0..127
  const int bh   = tid & 1;               // B loader 16-col half

  v8f acc[4][4];
  #pragma unroll
  for (int i = 0; i < 4; ++i)
    #pragma unroll
    for (int j = 0; j < 4; ++j)
      acc[i][j] = v8f{};

  // per-thread LDS byte addresses for both buffers (low 32 bits of flat addr)
  const uint32_t ldsA0 = (uint32_t)(uintptr_t)(&As[0][tid * LDS_PITCH]);
  const uint32_t ldsA1 = (uint32_t)(uintptr_t)(&As[1][tid * LDS_PITCH]);
  const uint32_t ldsB0 = (uint32_t)(uintptr_t)(&Bs[0][brow * LDS_PITCH + bh * 8]);
  const uint32_t ldsB1 = (uint32_t)(uintptr_t)(&Bs[1][brow * LDS_PITCH + bh * 8]);

  auto issue_tile = [&](int kt, int buf) {
    int k0 = kt * BK;                     // 32-aligned: never straddles HID
    const uint16_t* ga = (k0 < HID)
        ? aL + (size_t)(blockM + tid) * HID  + k0
        : aR + (size_t)(blockM + tid) * INPW + (k0 - HID);
    uint32_t la = buf ? ldsA1 : ldsA0;
    ASYNC_B128(la, ga, 0);
    ASYNC_B128(la, ga, 16);
    ASYNC_B128(la, ga, 32);
    ASYNC_B128(la, ga, 48);
    const uint16_t* gb = w16 + (size_t)(blockN + brow) * KTOT + k0 + bh * 16;
    uint32_t lb = buf ? ldsB1 : ldsB0;
    ASYNC_B128(lb, gb, 0);
    ASYNC_B128(lb, gb, 16);
  };

  issue_tile(0, 0);

  const int nK = KTOT / BK;               // 128
  #pragma clang loop unroll(disable)
  for (int kt = 0; kt < nK; ++kt) {
    const int cur = kt & 1;
    if (kt + 1 < nK) {
      issue_tile(kt + 1, cur ^ 1);        // 6 new ops stay in flight
      asm volatile("s_wait_asynccnt 6" ::: "memory");  // tile kt landed
    } else {
      asm volatile("s_wait_asynccnt 0" ::: "memory");
    }
    __syncthreads();                      // tile kt visible to all waves

    const uint32_t* asb = As[cur];
    const uint32_t* bsb = Bs[cur];

    FragU bfm[4];
    #pragma unroll
    for (int nt = 0; nt < 4; ++nt) {
      // B lane layout: col = fr; K = fh*16 .. +15 contiguous
      const uint32_t* p = bsb + (waveN + nt * 16 + fr) * LDS_PITCH + fh * 8;
      bfm[nt].u[0] = *reinterpret_cast<const uint4*>(p);
      bfm[nt].u[1] = *reinterpret_cast<const uint4*>(p + 4);
    }
    #pragma unroll
    for (int mt = 0; mt < 4; ++mt) {
      // A lane layout: row = fr; K = fh*16 .. +15 contiguous
      FragU af;
      const uint32_t* p = asb + (waveM + mt * 16 + fr) * LDS_PITCH + fh * 8;
      af.u[0] = *reinterpret_cast<const uint4*>(p);
      af.u[1] = *reinterpret_cast<const uint4*>(p + 4);
      #pragma unroll
      for (int nt = 0; nt < 4; ++nt)
        acc[mt][nt] = __builtin_amdgcn_wmma_f32_16x16x32_bf16(
            false, af.v, false, bfm[nt].v, (short)0, acc[mt][nt],
            false, false);
    }

    __syncthreads();                      // all reads of buf[cur] done
  }

  // Epilogue
  #pragma unroll
  for (int nt = 0; nt < 4; ++nt) {
    int n = blockN + waveN + nt * 16 + fr;
    float bias = bias_v[n];
    #pragma unroll
    for (int mt = 0; mt < 4; ++mt) {
      int mbase = blockM + waveM + mt * 16 + fh * 8;
      #pragma unroll
      for (int v = 0; v < 8; ++v) {
        int m = mbase + v;
        size_t off = (size_t)m * HID + n;
        if (PHASE == 1) {
          float fg = (acc[mt][nt][v] + bias + 1.0f) * 0.5f;
          float hv = hx[off];
          fg_ws[off] = fg;
          fgh16[off] = f32_to_bf16(fg * hv);
        } else {
          float ng = acc[mt][nt][v] + bias;
          float fg = fg_ws[off];
          float hv = hx[off];
          out[off] = (1.0f - fg) * ng + fg * hv;
        }
      }
    }
  }
}

// ---------------------------------------------------------------------------
extern "C" void kernel_launch(void* const* d_in, const int* in_sizes, int n_in,
                              void* d_out, int out_size, void* d_ws, size_t ws_size,
                              hipStream_t stream) {
  const float* inp = (const float*)d_in[0];
  const float* hx  = (const float*)d_in[1];
  const float* w_f = (const float*)d_in[2];
  const float* b_f = (const float*)d_in[3];
  const float* w_n = (const float*)d_in[4];
  const float* b_n = (const float*)d_in[5];
  float* out = (float*)d_out;

  // ws: fg f32 16MiB | hx16 8MiB | in16 8MiB | fgh16 8MiB | wf16 16MiB | wn16 16MiB
  char* ws = (char*)d_ws;
  float*    fg_ws = (float*)ws;          ws += (size_t)BATCH * HID * 4;
  uint16_t* hx16  = (uint16_t*)ws;       ws += (size_t)BATCH * HID * 2;
  uint16_t* in16  = (uint16_t*)ws;       ws += (size_t)BATCH * INPW * 2;
  uint16_t* fgh16 = (uint16_t*)ws;       ws += (size_t)BATCH * HID * 2;
  uint16_t* wf16  = (uint16_t*)ws;       ws += (size_t)HID * KTOT * 2;
  uint16_t* wn16  = (uint16_t*)ws;

  const int actElems = BATCH * HID;      // 4 M
  const int wElems   = HID * KTOT;       // 8 M
  mgu_cvt_bf16<<<actElems / 8 / 256, 256, 0, stream>>>(hx, hx16);
  mgu_cvt_bf16<<<actElems / 8 / 256, 256, 0, stream>>>(inp, in16);
  mgu_cvt_bf16<<<wElems / 8 / 256, 256, 0, stream>>>(w_f, wf16);
  mgu_cvt_bf16<<<wElems / 8 / 256, 256, 0, stream>>>(w_n, wn16);

  dim3 grid(HID / BN, BATCH / BM);       // 16 x 8
  mgu_gemm<1><<<grid, NTHREADS, 0, stream>>>(hx16, in16, wf16, b_f, hx,
                                             fg_ws, fgh16, out);
  mgu_gemm<2><<<grid, NTHREADS, 0, stream>>>(fgh16, in16, wn16, b_n, hx,
                                             fg_ws, fgh16, out);
}
